// NodeEmbedder_33904471835176
// MI455X (gfx1250) — compile-verified, matrix operand order checked by reading, not run on hardware
//
#include <hip/hip_runtime.h>
#include <hip/hip_bf16.h>

// ---- problem constants (from reference) ----
#define DIN  768
#define HDIM 128
#define DOUTC 64

typedef __bf16 bf16_t;
typedef __attribute__((ext_vector_type(4)))  __bf16 v4bf;
typedef __attribute__((ext_vector_type(8)))  __bf16 v8bf;
typedef __attribute__((ext_vector_type(16))) __bf16 v16bf;
typedef __attribute__((ext_vector_type(8)))  float  v8f;

// ------------------------------------------------------------------
// Small utility kernels
// ------------------------------------------------------------------
__global__ void k_zero_f32(float* __restrict__ p, int n) {
    int i = blockIdx.x * blockDim.x + threadIdx.x;
    if (i < n) p[i] = 0.0f;
}

__global__ void k_deg_count(const int* __restrict__ dst, float* __restrict__ deg, int E) {
    int i = blockIdx.x * blockDim.x + threadIdx.x;
    if (i < E) atomicAdd(&deg[dst[i]], 1.0f);
}

__global__ void k_dinv(const float* __restrict__ deg, float* __restrict__ dinv, int n) {
    int i = blockIdx.x * blockDim.x + threadIdx.x;
    if (i < n) dinv[i] = rsqrtf(deg[i] + 1.0f);
}

// h[i] = relu(h[i] + b[col])
__global__ void k_bias_relu(float* __restrict__ h, const float* __restrict__ b, int total) {
    int i = blockIdx.x * blockDim.x + threadIdx.x;
    if (i >= total) return;
    float v = h[i] + b[i & (HDIM - 1)];
    h[i] = v > 0.0f ? v : 0.0f;
}

// ------------------------------------------------------------------
// Edge scatter: one wave32 per edge; lane handles 4 consecutive floats
// (32 lanes * 4 = 128 = HDIM). Gathers t[src] (L2 resident, coalesced
// 512B per wave), atomically accumulates into agg[dst] via f32 L2 atomics.
// ------------------------------------------------------------------
__global__ void k_scatter(const int* __restrict__ src, const int* __restrict__ dst,
                          const float* __restrict__ dinv, const float* __restrict__ t,
                          float* __restrict__ agg, int E) {
    int wave = (blockIdx.x * blockDim.x + threadIdx.x) >> 5;
    int lane = threadIdx.x & 31;
    if (wave >= E) return;
    int s = src[wave];
    int d = dst[wave];
    float norm = dinv[s] * dinv[d];
    const float4* trow = (const float4*)(t + (size_t)s * HDIM);
    float4 v = trow[lane];
    float* out = agg + (size_t)d * HDIM + lane * 4;
    atomicAdd(out + 0, v.x * norm);
    atomicAdd(out + 1, v.y * norm);
    atomicAdd(out + 2, v.z * norm);
    atomicAdd(out + 3, v.w * norm);
}

// ------------------------------------------------------------------
// Pack a row-major f32 weight matrix W[K, Nout] into per-lane WMMA
// B-fragment order (ISA 7.12.2, 16-bit B 32x16):
//   lane l: col n = nt*16 + (l&15)
//   lanes<16 : K = kt*32+{0..7} then kt*32+{16..23}
//   lanes>=16: K = kt*32+{8..15} then kt*32+{24..31}
// Tile (kt,nt) stored at Bp + (kt*nTiles + nt)*32 lanes * 16 bf16.
// ------------------------------------------------------------------
__global__ void k_pack_w(const float* __restrict__ W, bf16_t* __restrict__ Bp,
                         int K, int Nout) {
    int nTiles = Nout >> 4;
    int kTiles = K >> 5;
    int tid = blockIdx.x * blockDim.x + threadIdx.x;
    int total = kTiles * nTiles * 32;
    if (tid >= total) return;
    int lane = tid & 31;
    int tile = tid >> 5;
    int nt = tile % nTiles;
    int kt = tile / nTiles;
    int n  = nt * 16 + (lane & 15);
    int k0 = kt * 32 + ((lane >> 4) << 3);
    bf16_t* out = Bp + (size_t)tid * 16;
#pragma unroll
    for (int j = 0; j < 8; ++j) out[j]     = (bf16_t)W[(size_t)(k0 + j) * Nout + n];
#pragma unroll
    for (int j = 0; j < 8; ++j) out[8 + j] = (bf16_t)W[(size_t)(k0 + 16 + j) * Nout + n];
}

// ------------------------------------------------------------------
// GEMM: C[M, Nout] = A[M, K] @ W (pre-packed bf16 B-fragments) [+ bias]
// Optionally also writes agg[row, col] = C[row, col] * dinv[row]^2
// (fused GCN self-loop term, saves a full L2 pass over t).
//
// One block per 16-row M tile; one wave per 16-col N tile.
// The 16xK A tile is staged to LDS ONCE per block, converted to bf16
// during staging (row stride K+8 bf16 -> 16B-aligned, bank-spread).
// K loop: ds_load_b128 fragments + v_wmma_f32_16x16x32_bf16.
// ------------------------------------------------------------------
__global__ void k_gemm_wmma(const float* __restrict__ A, const bf16_t* __restrict__ Bp,
                            float* __restrict__ C, const float* __restrict__ bias,
                            float* __restrict__ agg, const float* __restrict__ dinv,
                            int M, int K, int Nout) {
    extern __shared__ __align__(16) bf16_t smem[];   // 16 x (K+8) bf16

    const int lane    = threadIdx.x & 31;
    const int ntile   = threadIdx.x >> 5;
    const int nTiles  = blockDim.x >> 5;
    const int halfSel = lane >> 4;                 // 0: lanes 0-15, 1: lanes 16-31
    const int mBase   = blockIdx.x * 16;
    const int KP      = K + 8;                     // padded LDS row stride (bf16)

    // ---- cooperative stage: 16 x K f32 -> bf16 in LDS (coalesced float4) ----
    const int kq = K >> 2;                         // float4s per row
    for (int q = threadIdx.x; q < 16 * kq; q += blockDim.x) {
        int r  = q / kq;
        int c4 = q - r * kq;
        int rowSrc = mBase + r;
        if (rowSrc >= M) rowSrc = M - 1;
        const float* ap = A + (size_t)rowSrc * K + c4 * 4;
        __builtin_prefetch(ap + 1024, 0, 1);       // global_prefetch_b8
        float4 v = *(const float4*)ap;
        v4bf pv;
        pv[0] = (bf16_t)v.x; pv[1] = (bf16_t)v.y; pv[2] = (bf16_t)v.z; pv[3] = (bf16_t)v.w;
        *(v4bf*)(smem + (size_t)r * KP + c4 * 4) = pv;   // ds_store_b64
    }
    __syncthreads();

    // ---- K loop: build A fragment from LDS, packed B from global ----
    const bf16_t* srow = smem + (size_t)(lane & 15) * KP;
    const int kTiles = K >> 5;
    v8f acc = {};
    for (int kt = 0; kt < kTiles; ++kt) {
        const int kb = kt * 32 + halfSel * 8;
        v8bf c0 = *(const v8bf*)(srow + kb);        // ds_load_b128
        v8bf c1 = *(const v8bf*)(srow + kb + 16);   // ds_load_b128
        v16bf a;
#pragma unroll
        for (int j = 0; j < 8; ++j) { a[j] = c0[j]; a[j + 8] = c1[j]; }

        const v16bf b = *(const v16bf*)(Bp + (((size_t)kt * nTiles + ntile) * 32 + lane) * 16);

        acc = __builtin_amdgcn_wmma_f32_16x16x32_bf16(
            /*neg_a=*/false, a, /*neg_b=*/false, b,
            /*c_mod=*/(short)0, acc, /*reuse_a=*/false, /*reuse_b=*/false);
    }

    // ---- D-fragment store (ISA 7.12.2): VGPR r -> M = r + 8*halfSel, N = lane&15 ----
    const int col = (ntile << 4) + (lane & 15);
    const float bv = bias ? bias[col] : 0.0f;
    const int rBase = mBase + halfSel * 8;
#pragma unroll
    for (int r = 0; r < 8; ++r) {
        int row = rBase + r;
        if (row < M) {
            float val = acc[r] + bv;
            C[(size_t)row * Nout + col] = val;
            if (agg) {                              // fused self-loop init
                float dd = dinv[row];
                agg[(size_t)row * Nout + col] = val * dd * dd;
            }
        }
    }
}

// ------------------------------------------------------------------
// Host-side orchestration
// ------------------------------------------------------------------
extern "C" void kernel_launch(void* const* d_in, const int* in_sizes, int n_in,
                              void* d_out, int out_size, void* d_ws, size_t ws_size,
                              hipStream_t stream) {
    const float* x   = (const float*)d_in[0];
    const int*   ei  = (const int*)  d_in[1];
    const float* W0  = (const float*)d_in[2];  const float* b0  = (const float*)d_in[3];
    const float* W1  = (const float*)d_in[4];  const float* b1  = (const float*)d_in[5];
    const float* W2  = (const float*)d_in[6];  const float* b2  = (const float*)d_in[7];
    const float* Wo1 = (const float*)d_in[8];  const float* bo1 = (const float*)d_in[9];
    const float* Wo2 = (const float*)d_in[10]; const float* bo2 = (const float*)d_in[11];
    float* out = (float*)d_out;

    const int N = in_sizes[0] / DIN;
    const int E = in_sizes[1] / 2;
    const int* src = ei;
    const int* dst = ei + E;

    // ---- workspace carve-up (all offsets 32B-aligned for v16bf loads) ----
    const size_t nh = (size_t)N * HDIM;
    float* bufA = (float*)d_ws;            // [N, 128] GEMM output t
    float* bufB = bufA + nh;               // [N, 128] agg / h (ping-pong)
    float* deg  = bufB + nh;               // [N]
    float* dinv = deg + N;                 // [N]
    size_t wOff = ((size_t)(dinv + N - (float*)d_ws) * 4 + 255) & ~(size_t)255;
    bf16_t* w0p  = (bf16_t*)((char*)d_ws + wOff);
    bf16_t* w1p  = w0p  + (size_t)DIN  * HDIM;
    bf16_t* w2p  = w1p  + (size_t)HDIM * HDIM;
    bf16_t* wo1p = w2p  + (size_t)HDIM * HDIM;
    bf16_t* wo2p = wo1p + (size_t)HDIM * HDIM;

    const int mTiles  = (N + 15) / 16;
    const int totalNH = (int)nh;

    // ---- degree / norm ----
    k_zero_f32<<<(N + 255) / 256, 256, 0, stream>>>(deg, N);
    k_deg_count<<<(E + 255) / 256, 256, 0, stream>>>(dst, deg, E);
    k_dinv<<<(N + 255) / 256, 256, 0, stream>>>(deg, dinv, N);

    // ---- pack weights into WMMA B-fragment order ----
    {
        int t0 = (DIN / 32) * (HDIM / 16) * 32;
        k_pack_w<<<(t0 + 255) / 256, 256, 0, stream>>>(W0, w0p, DIN, HDIM);
        int t1 = (HDIM / 32) * (HDIM / 16) * 32;
        k_pack_w<<<(t1 + 255) / 256, 256, 0, stream>>>(W1,  w1p,  HDIM, HDIM);
        k_pack_w<<<(t1 + 255) / 256, 256, 0, stream>>>(W2,  w2p,  HDIM, HDIM);
        k_pack_w<<<(t1 + 255) / 256, 256, 0, stream>>>(Wo1, wo1p, HDIM, HDIM);
        int t2 = (HDIM / 32) * (DOUTC / 16) * 32;
        k_pack_w<<<(t2 + 255) / 256, 256, 0, stream>>>(Wo2, wo2p, HDIM, DOUTC);
    }

    const int gemmBlkH  = (HDIM / 16) * 32;          // 256 threads = 8 waves
    const int ldsH      = 16 * (DIN  + 8) * 2;       // layer-0 A tile (bf16)
    const int ldsS      = 16 * (HDIM + 8) * 2;       // H-dim A tile (bf16)
    const int ewBlocks  = (totalNH + 255) / 256;
    const int scBlocks  = (E * 32 + 255) / 256;

    // Layer 0: t = x@W0; agg = t*dinv^2 (fused); scatter; relu+bias
    k_gemm_wmma<<<mTiles, gemmBlkH, ldsH, stream>>>(x, w0p, bufA, nullptr, bufB, dinv, N, DIN, HDIM);
    k_scatter<<<scBlocks, 256, 0, stream>>>(src, dst, dinv, bufA, bufB, E);
    k_bias_relu<<<ewBlocks, 256, 0, stream>>>(bufB, b0, totalNH);

    // Layer 1
    k_gemm_wmma<<<mTiles, gemmBlkH, ldsS, stream>>>(bufB, w1p, bufA, nullptr, bufB, dinv, N, HDIM, HDIM);
    k_scatter<<<scBlocks, 256, 0, stream>>>(src, dst, dinv, bufA, bufB, E);
    k_bias_relu<<<ewBlocks, 256, 0, stream>>>(bufB, b1, totalNH);

    // Layer 2
    k_gemm_wmma<<<mTiles, gemmBlkH, ldsS, stream>>>(bufB, w2p, bufA, nullptr, bufB, dinv, N, HDIM, HDIM);
    k_scatter<<<scBlocks, 256, 0, stream>>>(src, dst, dinv, bufA, bufB, E);
    k_bias_relu<<<ewBlocks, 256, 0, stream>>>(bufB, b2, totalNH);

    // Head: h@Wo1 + bo1 (no relu), then @Wo2 + bo2 -> d_out [N, 64]
    k_gemm_wmma<<<mTiles, gemmBlkH, ldsS, stream>>>(bufB, wo1p, bufA, bo1, nullptr, nullptr, N, HDIM, HDIM);
    const int gemmBlkO = (DOUTC / 16) * 32;          // 128 threads = 4 waves
    k_gemm_wmma<<<mTiles, gemmBlkO, ldsS, stream>>>(bufA, wo2p, out, bo2, nullptr, nullptr, N, HDIM, DOUTC);

    (void)n_in; (void)out_size; (void)ws_size;
}
